// SelfAttention_10909216932392
// MI455X (gfx1250) — matrix operand order repaired and checked by reading, not compiled, not run on hardware
//
#include <hip/hip_runtime.h>
#include <hip/hip_bf16.h>

// MI455X / gfx1250 flash self-attention with V_WMMA_F32_16X16X32_F16.
// B=8, C=256, H=W=64 -> N=4096, Cqk=32 (== WMMA K for f16: perfect fit).

typedef _Float16 h16 __attribute__((ext_vector_type(16)));
typedef _Float16 h8  __attribute__((ext_vector_type(8)));
typedef float    f8  __attribute__((ext_vector_type(8)));
typedef float    f4  __attribute__((ext_vector_type(4)));

#define C_IN 256
#define NTOK 4096
#define CQK  32

__device__ __forceinline__ h16 combine(h8 lo, h8 hi) {
  h16 r;
  #pragma unroll
  for (int i = 0; i < 8; ++i) { r[i] = lo[i]; r[8 + i] = hi[i]; }
  return r;
}

__device__ __forceinline__ f8 wmma_f16(h16 a, h16 b, f8 c) {
  return __builtin_amdgcn_wmma_f32_16x16x32_f16(false, a, false, b,
                                                (short)0, c, false, false);
}

// ---------------------------------------------------------------------------
// Phase 1: QKV projections (1x1 convs) as WMMA GEMMs.
//  q  [b][n][d]          f16   (row-contiguous d: B-tile loads are 32B/lane)
//  kT [b][m][d]          f16   (A-tile loads are 2x16B/lane)
//  vS [b][m/16][c][16]   f16   (pre-swizzled so PV B-tile is 32B/lane)
// Grid: B * (N/64) blocks, 256 threads (8 waves).
// ---------------------------------------------------------------------------
#define XS_STRIDE 264  // 256 + 8 halves pad -> conflict-free 16B LDS reads

__global__ __launch_bounds__(256) void proj_kernel(
    const float* __restrict__ x,
    const float* __restrict__ Wq, const float* __restrict__ bq,
    const float* __restrict__ Wk, const float* __restrict__ bk,
    const float* __restrict__ Wv, const float* __restrict__ bv,
    _Float16* __restrict__ q, _Float16* __restrict__ kT,
    _Float16* __restrict__ vS)
{
  __shared__ _Float16 xS[64 * XS_STRIDE];  // x block transposed: [n][c], f16

  const int b  = blockIdx.x >> 6;   // 64 n-blocks of 64 per batch
  const int n0 = (blockIdx.x & 63) * 64;
  const int t  = threadIdx.x;

  // Stage x[c][n0..n0+63] -> xS[n][c] (f32->f16 once).
  const float* xb = x + (size_t)b * C_IN * NTOK;
  #pragma unroll 4
  for (int i = 0; i < 64; ++i) {
    int n = t & 63;
    int c = (t >> 6) + i * 4;
    xS[n * XS_STRIDE + c] = (_Float16)xb[(size_t)c * NTOK + n0 + n];
  }
  __syncthreads();

  const int lane  = t & 31;
  const int wave  = t >> 5;
  const int l16   = lane & 15;
  const int lhalf = lane >> 4;
  const int kbase = lhalf ? 8 : 0;   // A-matrix K interleave per half

  for (int j = wave; j < 80; j += 8) {
    if (j < 16) {
      // --- q / k:  D[o,n] = W[16o x 32c] * xS^T[32c x 16n], K-loop over c ---
      const float*   W    = (j < 8) ? Wq : Wk;
      const float*   bias = (j < 8) ? bq : bk;
      _Float16*      dst  = (j < 8) ? q  : kT;
      const int jj = j & 7;
      const int ot = jj >> 2;           // o-tile 0..1 (Cqk=32)
      const int nt = jj & 3;            // n-tile 0..3
      const int orow = ot * 16 + l16;
      const int ncol = nt * 16 + l16;

      f8 acc;
      #pragma unroll
      for (int r = 0; r < 8; ++r) acc[r] = bias[ot * 16 + 8 * lhalf + r];

      #pragma unroll
      for (int c0 = 0; c0 < C_IN; c0 += 32) {
        // A: weights row orow, K = {kbase..kbase+7, kbase+16..kbase+23}
        const float* wr = W + (size_t)orow * C_IN + c0 + kbase;
        h16 A;
        #pragma unroll
        for (int i2 = 0; i2 < 8; ++i2) {
          A[i2]     = (_Float16)wr[i2];
          A[8 + i2] = (_Float16)wr[16 + i2];
        }
        // B: xS[ncol][c0 + kbase .. +7, c0 + kbase + 16 .. +23] (16B-aligned LDS reads)
        const h8* xr = (const h8*)&xS[ncol * XS_STRIDE + c0 + kbase];
        h16 Bm = combine(xr[0], xr[2]);
        acc = wmma_f16(A, Bm, acc);
      }
      // lane holds column n, rows o = ot*16 + 8*lhalf + r (contiguous f16)
      h8 st;
      #pragma unroll
      for (int r = 0; r < 8; ++r) st[r] = (_Float16)acc[r];
      *(h8*)(dst + ((size_t)b * NTOK + n0 + ncol) * CQK + ot * 16 + 8 * lhalf) = st;
    } else {
      // --- v (as V^T): D[m,cv] = xS[16m x 32c] * Wv^T[32c x 16cv] ---
      const int jv  = j - 16;
      const int mt  = jv & 3;           // m-tile 0..3 within block
      const int cvt = jv >> 2;          // c-tile 0..15
      const int mrow = mt * 16 + l16;
      const int cv   = cvt * 16 + l16;

      f8 acc;
      { float bvv = bv[cv];
        #pragma unroll
        for (int r = 0; r < 8; ++r) acc[r] = bvv; }

      #pragma unroll
      for (int c0 = 0; c0 < C_IN; c0 += 32) {
        const h8* xr0 = (const h8*)&xS[mrow * XS_STRIDE + c0 + 16 * lhalf];
        const h8* xr1 = (const h8*)&xS[mrow * XS_STRIDE + c0 + 16 * lhalf + 8];
        h16 A = combine(xr0[0], xr1[0]);
        const float* wr = Wv + (size_t)cv * C_IN + c0 + 16 * lhalf;
        h16 Bm;
        #pragma unroll
        for (int i2 = 0; i2 < 16; ++i2) Bm[i2] = (_Float16)wr[i2];
        acc = wmma_f16(A, Bm, acc);
      }
      // lane holds column cv, rows m = base + 8*lhalf + r -> contiguous mlo
      const int tGlob = (n0 >> 4) + mt;
      h8 st;
      #pragma unroll
      for (int r = 0; r < 8; ++r) st[r] = (_Float16)acc[r];
      *(h8*)(vS + (((size_t)b * (NTOK / 16) + tGlob) * C_IN + cv) * 16
                + 8 * lhalf) = st;
    }
  }
}

// ---------------------------------------------------------------------------
// Phase 2: flash attention + gamma*out + x.
// Wave = one 16-row n-block x 128 channels (8 c-tiles of accumulators).
// Per m-step(32): 2 WMMAs for S^T, online softmax, 8 WMMAs for O += P*V.
// S^T trick: exp'd probabilities land exactly in the A-register layout.
// Grid: B*(N/16)/4 blocks, 256 threads (8 waves; pairs split c 0..127/128..255).
// ---------------------------------------------------------------------------
__global__ __launch_bounds__(256) void attn_kernel(
    const _Float16* __restrict__ q, const _Float16* __restrict__ kT,
    const _Float16* __restrict__ vS, const float* __restrict__ x,
    const float* __restrict__ gamma, float* __restrict__ out)
{
  const int lane  = threadIdx.x & 31;
  const int wave  = threadIdx.x >> 5;
  const int l16   = lane & 15;
  const int lhalf = lane >> 4;

  const int nbG   = blockIdx.x * 4 + (wave >> 1);  // global n-block 0..2047
  const int b     = nbG >> 8;                      // 256 n-blocks per batch
  const int nb    = nbG & 255;
  const int chalf = wave & 1;

  const _Float16* qb = q  + (size_t)b * NTOK * CQK;
  const _Float16* kb = kT + (size_t)b * NTOK * CQK;
  const _Float16* vb = vS + (size_t)b * (NTOK / 16) * C_IN * 16;

  // Loop-invariant B-tile: q^T [32d x 16n]; lane = column n, halves split d.
  h16 Bq;
  {
    const h8* p = (const h8*)(qb + ((size_t)nb * 16 + l16) * CQK + 8 * lhalf);
    Bq = combine(p[0], p[2]);
  }

  f8 O[8];
  #pragma unroll
  for (int j = 0; j < 8; ++j)
    #pragma unroll
    for (int r = 0; r < 8; ++r) O[j][r] = 0.f;

  float mi = -1e30f;   // running row max (per lane's column n)
  float li = 0.f;      // running row sum

  for (int m0 = 0; m0 < NTOK; m0 += 32) {
    // A-tiles: k^T rows m0..m0+31, interleaved K per lane-half.
    const _Float16* k0 = kb + ((size_t)m0 + l16) * CQK + (lhalf ? 8 : 0);
    const _Float16* k1 = k0 + 16 * CQK;
    h16 A0 = combine(*(const h8*)k0, *(const h8*)(k0 + 16));
    h16 A1 = combine(*(const h8*)k1, *(const h8*)(k1 + 16));

    // Prefetch next k tile into cache (global_prefetch_b8).
    __builtin_prefetch(k0 + 32 * CQK, 0, 1);

    f8 z = {};
    f8 S0 = wmma_f16(A0, Bq, z);   // S^T[m0..m0+15 , n]
    f8 S1 = wmma_f16(A1, Bq, z);   // S^T[m0+16..m0+31, n]

    // Online softmax over m (lane holds 16 m-values for one n; finish with
    // one xor-16 shuffle across the two lane halves).
    float mloc = S0[0];
    #pragma unroll
    for (int r = 1; r < 8; ++r) mloc = fmaxf(mloc, S0[r]);
    #pragma unroll
    for (int r = 0; r < 8; ++r) mloc = fmaxf(mloc, S1[r]);
    mloc = fmaxf(mloc, __shfl_xor(mloc, 16, 32));
    float mnew  = fmaxf(mi, mloc);
    float alpha = __expf(mi - mnew);

    // P lands directly in A-layout: P[0..7]=tile0 rows, P[8..15]=tile1 rows.
    h16 P;
    float ps = 0.f;
    #pragma unroll
    for (int r = 0; r < 8; ++r) {
      float p = __expf(S0[r] - mnew); ps += p; P[r] = (_Float16)p;
    }
    #pragma unroll
    for (int r = 0; r < 8; ++r) {
      float p = __expf(S1[r] - mnew); ps += p; P[8 + r] = (_Float16)p;
    }
    ps += __shfl_xor(ps, 16, 32);
    li = li * alpha + ps;
    mi = mnew;

    // Rescale O: O-tile rows are n = 8*lhalf + r -> broadcast alpha from the
    // lane owning that column.
    float ab[8];
    #pragma unroll
    for (int r = 0; r < 8; ++r) ab[r] = __shfl(alpha, 8 * lhalf + r, 32);
    #pragma unroll
    for (int j = 0; j < 8; ++j)
      #pragma unroll
      for (int r = 0; r < 8; ++r) O[j][r] *= ab[r];

    // O += P * V : B-tile from swizzled vS, 2x16B per lane (tiles t0, t0+1).
    const int t0 = m0 >> 4;
    #pragma unroll
    for (int j = 0; j < 8; ++j) {
      const int cc = chalf * 128 + j * 16 + l16;
      const h8* vp = (const h8*)(vb + (((size_t)t0) * C_IN + cc) * 16 + 8 * lhalf);
      __builtin_prefetch(vp + 64 * C_IN / 8, 0, 1);  // v block two m-steps ahead (tile t0+4)
      h16 Bv = combine(vp[0], vp[C_IN * 2]);
      O[j] = wmma_f16(P, Bv, O[j]);
    }
  }

  // Epilogue: out = gamma * O / l + x.
  float lb[8];
  #pragma unroll
  for (int r = 0; r < 8; ++r) lb[r] = __shfl(li, 8 * lhalf + r, 32);
  const float g = gamma[0];
  const float* xb = x   + (size_t)b * C_IN * NTOK;
  float*       ob = out + (size_t)b * C_IN * NTOK;
  #pragma unroll
  for (int j = 0; j < 8; ++j) {
    const int cc = chalf * 128 + j * 16 + l16;
    const size_t off = (size_t)cc * NTOK + (size_t)nb * 16 + 8 * lhalf;
    f4 xv0 = *(const f4*)(xb + off);
    f4 xv1 = *(const f4*)(xb + off + 4);
    f4 o0, o1;
    #pragma unroll
    for (int r = 0; r < 4; ++r) {
      o0[r] = g * O[j][r]     / lb[r]     + xv0[r];
      o1[r] = g * O[j][4 + r] / lb[4 + r] + xv1[r];
    }
    *(f4*)(ob + off)     = o0;
    *(f4*)(ob + off + 4) = o1;
  }
}

// ---------------------------------------------------------------------------
extern "C" void kernel_launch(void* const* d_in, const int* in_sizes, int n_in,
                              void* d_out, int out_size, void* d_ws,
                              size_t ws_size, hipStream_t stream) {
  (void)in_sizes; (void)n_in; (void)out_size; (void)ws_size;
  const float* x     = (const float*)d_in[0];
  const float* Wq    = (const float*)d_in[1];
  const float* bq    = (const float*)d_in[2];
  const float* Wk    = (const float*)d_in[3];
  const float* bk    = (const float*)d_in[4];
  const float* Wv    = (const float*)d_in[5];
  const float* bv    = (const float*)d_in[6];
  const float* gamma = (const float*)d_in[7];

  // Workspace: q (2MB f16) | kT (2MB f16) | vS (16MB f16)  => 20MB total.
  _Float16* q  = (_Float16*)d_ws;
  _Float16* kT = q  + (size_t)8 * NTOK * CQK;
  _Float16* vS = kT + (size_t)8 * NTOK * CQK;

  proj_kernel<<<8 * (NTOK / 64), 256, 0, stream>>>(x, Wq, bq, Wk, bk, Wv, bv,
                                                   q, kT, vS);
  attn_kernel<<<8 * (NTOK / 16) / 4, 256, 0, stream>>>(q, kT, vS, x, gamma,
                                                       (float*)d_out);
}
